// GraphEncoder_64914135711785
// MI455X (gfx1250) — matrix operand order repaired and checked by reading, not compiled, not run on hardware
//
#include <hip/hip_runtime.h>
#include <math.h>

#define N_NODES 50000
#define N_EDGES 800000
#define IN_DIM  256
#define HC      128
#define HEADS   4
#define CH      32

typedef __attribute__((ext_vector_type(16))) __bf16 v16bf;
typedef __attribute__((ext_vector_type(8)))  float  v8f;

// ---------- order-preserving uint encoding for float atomic max ----------
__device__ __forceinline__ unsigned ordf(float f) {
  unsigned u = __float_as_uint(f);
  return (u & 0x80000000u) ? ~u : (u | 0x80000000u);
}
__device__ __forceinline__ float unordf(unsigned x) {
  unsigned u = (x & 0x80000000u) ? (x & 0x7FFFFFFFu) : ~x;
  return __uint_as_float(u);
}
// ---------- bf16 <-> f32 (shift based, RNE on pack) ----------
__device__ __forceinline__ unsigned short f2bf(float f) {
  unsigned u = __float_as_uint(f);
  u += 0x7FFFu + ((u >> 16) & 1u);              // round-to-nearest-even
  return (unsigned short)(u >> 16);
}
__device__ __forceinline__ float bfraw2f(unsigned short v) {
  return __uint_as_float(((unsigned)v) << 16);
}
__device__ __forceinline__ float bflo(unsigned u) { return __uint_as_float(u << 16); }
__device__ __forceinline__ float bfhi(unsigned u) { return __uint_as_float(u & 0xFFFF0000u); }

// =========================================================================
// GEMM: C[M,128] = A[M,K] @ W[K,128] (+ bias). fp32 in, bf16 WMMA core,
// fp32 OR bf16 out. Block = 256 threads = 8 waves, computes 64 rows x 128
// cols. Each wave owns one 16-col tile and 4 row tiles (4 accumulators):
// one B fragment feeds 4 WMMAs per K-step.
// A tile is CONVERTED TO BF16 AT STAGE TIME (once per block, not per wave):
// LDS rows padded to 40 bf16 (80 B = 20 banks -> conflict-free, uint4
// aligned). Each A fragment is then exactly two ds_load_b128, bit-cast into
// the v16bf WMMA operand -- no conversions in the inner loop.
// =========================================================================
__global__ void gemm_bias_wmma(const float* __restrict__ A,
                               const float* __restrict__ W,
                               const float* __restrict__ bias,
                               float* __restrict__ Cf,
                               unsigned short* __restrict__ Cbf,
                               int M, int K)
{
  __shared__ __align__(16) unsigned short lds[64 * 40];  // bf16 tile, 80B rows
  const int tid  = threadIdx.x;
  const int lane = tid & 31;
  const int wave = tid >> 5;                  // 0..7 -> column tile
  const int half = lane >> 4;                 // 0/1
  const int m    = lane & 15;
  const int row0 = blockIdx.x * 64;
  const int ncol = wave * 16 + m;             // output column (0..127)

  v8f acc[4] = {{}, {}, {}, {}};

  for (int k0 = 0; k0 < K; k0 += 32) {
    __syncthreads();
    {
      // stage 64x32 A tile, converting fp32 -> bf16 once for the block
      int r = tid >> 2;                       // 0..63
      int c = (tid & 3) * 8;                  // 0,8,16,24
      int rg = row0 + r; if (rg > M - 1) rg = M - 1;   // clamp (loads only)
      const float4* ap = (const float4*)(A + (size_t)rg * K + k0 + c);
      float4 a0 = ap[0], a1 = ap[1];
      union { __bf16 h[8]; uint4 u; } pk;
      pk.h[0] = (__bf16)a0.x; pk.h[1] = (__bf16)a0.y;
      pk.h[2] = (__bf16)a0.z; pk.h[3] = (__bf16)a0.w;
      pk.h[4] = (__bf16)a1.x; pk.h[5] = (__bf16)a1.y;
      pk.h[6] = (__bf16)a1.z; pk.h[7] = (__bf16)a1.w;
      ((uint4*)lds)[r * 5 + (c >> 3)] = pk.u;     // byte off = r*80 + c*2
    }
    __syncthreads();

    // B fragment (32x16, K = 8*half + (i&7) + 16*(i>>3)), reused by all 4 row tiles
    v16bf bfv;
    #pragma unroll
    for (int i = 0; i < 16; ++i)
      bfv[i] = (__bf16)W[(size_t)(k0 + half * 8 + (i & 7) + (i >> 3) * 16) * HC + ncol];

    const uint4* lu = (const uint4*)lds;
    #pragma unroll
    for (int rt = 0; rt < 4; ++rt) {
      // A frag: elements 0..7  <- K = half*8 + 0..7      (16B run)
      //         elements 8..15 <- K = 16 + half*8 + 0..7 (16B run)
      int row = rt * 16 + m;
      union { v16bf v; uint4 u[2]; } af;
      af.u[0] = lu[row * 5 + half];
      af.u[1] = lu[row * 5 + 2 + half];
      acc[rt] = __builtin_amdgcn_wmma_f32_16x16x32_bf16(
          false, af.v, false, bfv, (short)0, acc[rt], false, false);
    }
  }

  const float bv = bias ? bias[ncol] : 0.0f;
  #pragma unroll
  for (int rt = 0; rt < 4; ++rt) {
    #pragma unroll
    for (int r = 0; r < 8; ++r) {
      int mrow = row0 + rt * 16 + r + half * 8;   // C layout: M = r + 8*half
      if (mrow < M) {
        float val = acc[rt][r] + bv;
        if (Cbf) Cbf[(size_t)mrow * HC + ncol] = f2bf(val);
        else     Cf [(size_t)mrow * HC + ncol] = val;
      }
    }
  }
}

// =========================================================================
// alpha[e,h] = <q[dst], k[src]+ef> / sqrt(32); atomic segment max.
// q/k via float4, ef via uint4 (8 packed bf16 per load).
// =========================================================================
__global__ void edge_alpha_kernel(const float* __restrict__ q,
                                  const float* __restrict__ k,
                                  const unsigned short* __restrict__ ef,
                                  const int* __restrict__ src,
                                  const int* __restrict__ dst,
                                  float* __restrict__ aRaw,
                                  unsigned* __restrict__ amax)
{
  int t = blockIdx.x * blockDim.x + threadIdx.x;
  if (t >= N_EDGES * HEADS) return;
  int e = t >> 2, h = t & 3;
  int s = src[e], d = dst[e];
  const float4* qp = (const float4*)(q + (size_t)d * HC + h * CH);
  const float4* kp = (const float4*)(k + (size_t)s * HC + h * CH);
  const uint4*  ep = (const uint4*)(ef + (size_t)e * HC + h * CH);
  float acc = 0.0f;
  #pragma unroll
  for (int j = 0; j < 4; ++j) {
    uint4  eu = ep[j];
    float4 q0 = qp[2 * j], q1 = qp[2 * j + 1];
    float4 k0 = kp[2 * j], k1 = kp[2 * j + 1];
    acc = fmaf(q0.x, k0.x + bflo(eu.x), acc);
    acc = fmaf(q0.y, k0.y + bfhi(eu.x), acc);
    acc = fmaf(q0.z, k0.z + bflo(eu.y), acc);
    acc = fmaf(q0.w, k0.w + bfhi(eu.y), acc);
    acc = fmaf(q1.x, k1.x + bflo(eu.z), acc);
    acc = fmaf(q1.y, k1.y + bfhi(eu.z), acc);
    acc = fmaf(q1.z, k1.z + bflo(eu.w), acc);
    acc = fmaf(q1.w, k1.w + bfhi(eu.w), acc);
  }
  acc *= 0.17677669529663688f;                 // 1/sqrt(32)
  aRaw[t] = acc;
  atomicMax(amax + (size_t)d * HEADS + h, ordf(acc));
}

// in place: a = exp(a - amax[dst]); atomic den accumulate
__global__ void edge_exp_kernel(float* __restrict__ a,
                                const unsigned* __restrict__ amax,
                                const int* __restrict__ dst,
                                float* __restrict__ den)
{
  int t = blockIdx.x * blockDim.x + threadIdx.x;
  if (t >= N_EDGES * HEADS) return;
  int e = t >> 2, h = t & 3;
  int d = dst[e];
  float mx = unordf(amax[(size_t)d * HEADS + h]);
  float ex = __expf(a[t] - mx);
  a[t] = ex;
  atomicAdd(den + (size_t)d * HEADS + h, ex);
}

// outacc[dst,c] += ex[e,h] * (v[src,c] + ef[e,c]);  one thread per (e,c)
__global__ void edge_scatter_kernel(const float* __restrict__ ex,
                                    const float* __restrict__ v,
                                    const unsigned short* __restrict__ ef,
                                    const int* __restrict__ src,
                                    const int* __restrict__ dst,
                                    float* __restrict__ outacc)
{
  int t = blockIdx.x * blockDim.x + threadIdx.x;
  int e = t >> 7, c = t & 127;
  if (e >= N_EDGES) return;
  int h = c >> 5;
  float a  = ex[(size_t)e * HEADS + h];
  float vj = v[(size_t)src[e] * HC + c] + bfraw2f(ef[(size_t)e * HC + c]);
  atomicAdd(outacc + (size_t)dst[e] * HC + c, a * vj);
}

// a[e,h] = ex[e,h] / (den[dst,h] + 1e-16)   (in place in d_out)
__global__ void alpha_norm_kernel(float* __restrict__ a,
                                  const float* __restrict__ den,
                                  const int* __restrict__ dst)
{
  int t = blockIdx.x * blockDim.x + threadIdx.x;
  if (t >= N_EDGES * HEADS) return;
  int e = t >> 2, h = t & 3;
  a[t] = a[t] / (den[(size_t)dst[e] * HEADS + h] + 1e-16f);
}

__device__ __forceinline__ float wave_sum(float v) {
  #pragma unroll
  for (int off = 16; off > 0; off >>= 1) v += __shfl_xor(v, off, 32);
  return v;
}

// per node: out = acc/(den+eps) + skip;  optional LayerNorm + ReLU.
// one wave32 per node, 4 channels per lane (c = j*32 + lane).
__global__ void finalize_kernel(const float* __restrict__ acc,
                                const float* __restrict__ den,
                                const float* __restrict__ skip,
                                const float* __restrict__ gamma,
                                const float* __restrict__ beta,
                                float* __restrict__ out,
                                int doLN)
{
  int lane = threadIdx.x & 31;
  int node = blockIdx.x * (blockDim.x >> 5) + (threadIdx.x >> 5);
  if (node >= N_NODES) return;
  float o[4];
  #pragma unroll
  for (int j = 0; j < 4; ++j) {
    int c = j * 32 + lane;
    float dd = den[(size_t)node * HEADS + j];
    o[j] = acc[(size_t)node * HC + c] / (dd + 1e-16f) + skip[(size_t)node * HC + c];
  }
  if (doLN) {
    float s = wave_sum(o[0] + o[1] + o[2] + o[3]);
    float mean = s * (1.0f / 128.0f);
    float vs = 0.0f;
    #pragma unroll
    for (int j = 0; j < 4; ++j) { float t = o[j] - mean; vs += t * t; }
    vs = wave_sum(vs);
    float inv = rsqrtf(vs * (1.0f / 128.0f) + 1e-5f);
    #pragma unroll
    for (int j = 0; j < 4; ++j) {
      int c = j * 32 + lane;
      float r = (o[j] - mean) * inv * gamma[c] + beta[c];
      out[(size_t)node * HC + c] = fmaxf(r, 0.0f);
    }
  } else {
    #pragma unroll
    for (int j = 0; j < 4; ++j) {
      int c = j * 32 + lane;
      out[(size_t)node * HC + c] = o[j];
    }
  }
}

// =========================================================================
extern "C" void kernel_launch(void* const* d_in, const int* in_sizes, int n_in,
                              void* d_out, int out_size, void* d_ws, size_t ws_size,
                              hipStream_t stream)
{
  const float* x    = (const float*)d_in[0];
  const int*   ei   = (const int*)d_in[1];
  const float* attr = (const float*)d_in[2];
  const int* src = ei;
  const int* dst = ei + N_EDGES;

  // per-layer weights: base 3 + 9*l : Wq,bq,Wk,bk,Wv,bv,We,Ws,bs
  const float *Wq[3], *bq[3], *Wk[3], *bk[3], *Wv[3], *bv[3], *We[3], *Ws[3], *bs[3];
  for (int l = 0; l < 3; ++l) {
    int b = 3 + 9 * l;
    Wq[l] = (const float*)d_in[b + 0]; bq[l] = (const float*)d_in[b + 1];
    Wk[l] = (const float*)d_in[b + 2]; bk[l] = (const float*)d_in[b + 3];
    Wv[l] = (const float*)d_in[b + 4]; bv[l] = (const float*)d_in[b + 5];
    We[l] = (const float*)d_in[b + 6];
    Ws[l] = (const float*)d_in[b + 7]; bs[l] = (const float*)d_in[b + 8];
  }
  const float* gam[2] = { (const float*)d_in[30], (const float*)d_in[32] };
  const float* bet[2] = { (const float*)d_in[31], (const float*)d_in[33] };

  float* out  = (float*)d_out;
  float* aOut[3];
  aOut[0] = out + (size_t)N_NODES * HC;
  aOut[1] = aOut[0] + (size_t)N_EDGES * HEADS;
  aOut[2] = aOut[1] + (size_t)N_EDGES * HEADS;

  // workspace carve-up
  const size_t NH = (size_t)N_NODES * HC;
  float* ws      = (float*)d_ws;
  float* qbuf    = ws;
  float* kbuf    = qbuf + NH;
  float* vbuf    = kbuf + NH;
  float* sbuf    = vbuf + NH;
  float* hbuf    = sbuf + NH;
  float* outacc  = hbuf + NH;                       // N*128  --+
  float* den     = outacc + NH;                     // N*4      | one memset
  unsigned* amax = (unsigned*)(den + (size_t)N_NODES * HEADS); // N*4 --+
  unsigned short* ef = (unsigned short*)(amax + (size_t)N_NODES * HEADS); // E*128 bf16

  const dim3 blk(256);
  const int gemmNodeGrid = (N_NODES + 63) / 64;     // 782
  const int gemmEdgeGrid = N_EDGES / 64;            // 12500
  const int ehGrid   = (N_EDGES * HEADS) / 256;     // 12500
  const int scatGrid = (N_EDGES * HC) / 256;        // 400000
  const int finGrid  = (N_NODES + 7) / 8;           // 6250

  for (int l = 0; l < 3; ++l) {
    const float* X = (l == 0) ? x : hbuf;
    int K = (l == 0) ? IN_DIM : HC;

    // clear outacc + den + amax (contiguous; amax init 0 == ordered -inf)
    hipMemsetAsync(outacc, 0, (NH + 2 * (size_t)N_NODES * HEADS) * sizeof(float), stream);

    // dense projections (bf16 WMMA, f32 accumulate)
    gemm_bias_wmma<<<gemmNodeGrid, blk, 0, stream>>>(X, Wq[l], bq[l], qbuf, nullptr, N_NODES, K);
    gemm_bias_wmma<<<gemmNodeGrid, blk, 0, stream>>>(X, Wk[l], bk[l], kbuf, nullptr, N_NODES, K);
    gemm_bias_wmma<<<gemmNodeGrid, blk, 0, stream>>>(X, Wv[l], bv[l], vbuf, nullptr, N_NODES, K);
    gemm_bias_wmma<<<gemmNodeGrid, blk, 0, stream>>>(X, Ws[l], bs[l], sbuf, nullptr, N_NODES, K);
    // edge features stored as bf16 (halves HBM traffic for the E x 128 tensor)
    gemm_bias_wmma<<<gemmEdgeGrid, blk, 0, stream>>>(attr, We[l], nullptr, nullptr, ef, N_EDGES, 32);

    // attention: raw alpha straight into d_out slot, exp in place
    edge_alpha_kernel<<<ehGrid, blk, 0, stream>>>(qbuf, kbuf, ef, src, dst, aOut[l], amax);
    edge_exp_kernel<<<ehGrid, blk, 0, stream>>>(aOut[l], amax, dst, den);
    edge_scatter_kernel<<<scatGrid, blk, 0, stream>>>(aOut[l], vbuf, ef, src, dst, outacc);

    // node finalize: + skip, then LN+ReLU for layers 0,1; raw for layer 2
    if (l < 2) {
      finalize_kernel<<<finGrid, blk, 0, stream>>>(outacc, den, sbuf,
                                                   gam[l], bet[l], hbuf, 1);
    } else {
      finalize_kernel<<<finGrid, blk, 0, stream>>>(outacc, den, sbuf,
                                                   nullptr, nullptr, out, 0);
    }
    // normalize attention weights in place (after scatter consumed raw ex)
    alpha_norm_kernel<<<ehGrid, blk, 0, stream>>>(aOut[l], den, dst);
  }
}
